// EnhancedTransformer_58909771432435
// MI455X (gfx1250) — matrix operand order repaired and checked by reading, not compiled, not run on hardware
//
#include <hip/hip_runtime.h>
#include <hip/hip_bf16.h>
#include <math.h>

// ---------------------------------------------------------------------------
// Problem constants (from reference)
// ---------------------------------------------------------------------------
#define NN   10000
#define EE   160000
#define FF   128
#define D1   512     // H1*C1 = 8*64
#define H1H  8
#define C1C  64
#define C2C  64
#define BB   64

typedef __attribute__((ext_vector_type(16))) __bf16 v16bf;
typedef __attribute__((ext_vector_type(8)))  float  v8f;

// Native f32 -> bf16 conversion (RNE); lowers to gfx1250 v_cvt_pk_bf16_f32
// instead of the 3-VALU-op bit-twiddle sequence.
__device__ __forceinline__ __bf16 f2bf(float f) {
  return static_cast<__bf16>(f);
}

__device__ __forceinline__ void atomicMaxFloat(float* addr, float val) {
  if (val >= 0.0f) {
    atomicMax(reinterpret_cast<int*>(addr), __float_as_int(val));
  } else {
    atomicMin(reinterpret_cast<unsigned int*>(addr), __float_as_uint(val));
  }
}

// ---------------------------------------------------------------------------
// GEMM: C[M,Nc] = A[M,K] (f32, row-major) x B[K,Nc] (f32, row-major) + bias
// bf16 WMMA, f32 accumulate. One wave (32 threads) per block.
// Requires M%16==0, K%32==0, Nc%64==0. grid = (Nc/64, M/16).
// ---------------------------------------------------------------------------
__global__ void __launch_bounds__(32)
gemm_bias_wmma_bf16(const float* __restrict__ A, const float* __restrict__ B,
                    const float* __restrict__ bias, float* __restrict__ C,
                    int M, int K, int Nc) {
  const int lane = threadIdx.x;
  const int m0   = blockIdx.y * 16;
  const int n0   = blockIdx.x * 64;
  const int koff = (lane < 16) ? 0 : 8;     // A-fragment K split per ISA layout
  const int mrow = m0 + (lane & 15);

  v8f acc0 = {}, acc1 = {}, acc2 = {}, acc3 = {};

  for (int kb = 0; kb < K; kb += 32) {
    // ---- A fragment (16x32 bf16): two contiguous 8-float chunks per lane ----
    const float* Arow = A + (size_t)mrow * K + kb;
    float4 a0 = *reinterpret_cast<const float4*>(Arow + koff);
    float4 a1 = *reinterpret_cast<const float4*>(Arow + koff + 4);
    float4 a2 = *reinterpret_cast<const float4*>(Arow + 16 + koff);
    float4 a3 = *reinterpret_cast<const float4*>(Arow + 16 + koff + 4);
    v16bf afrag;
    afrag[0]  = f2bf(a0.x); afrag[1]  = f2bf(a0.y);
    afrag[2]  = f2bf(a0.z); afrag[3]  = f2bf(a0.w);
    afrag[4]  = f2bf(a1.x); afrag[5]  = f2bf(a1.y);
    afrag[6]  = f2bf(a1.z); afrag[7]  = f2bf(a1.w);
    afrag[8]  = f2bf(a2.x); afrag[9]  = f2bf(a2.y);
    afrag[10] = f2bf(a2.z); afrag[11] = f2bf(a2.w);
    afrag[12] = f2bf(a3.x); afrag[13] = f2bf(a3.y);
    afrag[14] = f2bf(a3.z); afrag[15] = f2bf(a3.w);

    // ---- B fragments (32x16 bf16): lane L = N-col L%16, K rows as A fragment ----
    const float* Brow = B + (size_t)(kb + koff) * Nc + n0 + (lane & 15);
#define LOAD_B_TILE(NT, FRAG)                                               \
    {                                                                       \
      const float* p = Brow + (NT) * 16;                                    \
      float4 b0 = make_float4(p[0], p[(size_t)Nc], p[(size_t)2 * Nc], p[(size_t)3 * Nc]); \
      float4 b1 = make_float4(p[(size_t)4 * Nc], p[(size_t)5 * Nc], p[(size_t)6 * Nc], p[(size_t)7 * Nc]); \
      float4 b2 = make_float4(p[(size_t)16 * Nc], p[(size_t)17 * Nc], p[(size_t)18 * Nc], p[(size_t)19 * Nc]); \
      float4 b3 = make_float4(p[(size_t)20 * Nc], p[(size_t)21 * Nc], p[(size_t)22 * Nc], p[(size_t)23 * Nc]); \
      FRAG[0]  = f2bf(b0.x); FRAG[1]  = f2bf(b0.y);                         \
      FRAG[2]  = f2bf(b0.z); FRAG[3]  = f2bf(b0.w);                         \
      FRAG[4]  = f2bf(b1.x); FRAG[5]  = f2bf(b1.y);                         \
      FRAG[6]  = f2bf(b1.z); FRAG[7]  = f2bf(b1.w);                         \
      FRAG[8]  = f2bf(b2.x); FRAG[9]  = f2bf(b2.y);                         \
      FRAG[10] = f2bf(b2.z); FRAG[11] = f2bf(b2.w);                         \
      FRAG[12] = f2bf(b3.x); FRAG[13] = f2bf(b3.y);                         \
      FRAG[14] = f2bf(b3.z); FRAG[15] = f2bf(b3.w);                         \
    }
    v16bf bfrag;
    LOAD_B_TILE(0, bfrag);
    acc0 = __builtin_amdgcn_wmma_f32_16x16x32_bf16(false, afrag, false, bfrag,
                                                   (short)0, acc0, false, false);
    LOAD_B_TILE(1, bfrag);
    acc1 = __builtin_amdgcn_wmma_f32_16x16x32_bf16(false, afrag, false, bfrag,
                                                   (short)0, acc1, false, false);
    LOAD_B_TILE(2, bfrag);
    acc2 = __builtin_amdgcn_wmma_f32_16x16x32_bf16(false, afrag, false, bfrag,
                                                   (short)0, acc2, false, false);
    LOAD_B_TILE(3, bfrag);
    acc3 = __builtin_amdgcn_wmma_f32_16x16x32_bf16(false, afrag, false, bfrag,
                                                   (short)0, acc3, false, false);
#undef LOAD_B_TILE
  }

  // ---- store: C/D layout VGPR r -> M = mbase+r, N = lane&15 ----
  const int ncl   = lane & 15;
  const int mbase = m0 + ((lane >> 4) << 3);
#define STORE_TILE(NT, ACCV)                                                \
  {                                                                         \
    int ncol = n0 + (NT) * 16 + ncl;                                        \
    float bv = bias[ncol];                                                  \
    _Pragma("unroll") for (int r = 0; r < 8; ++r)                           \
        C[(size_t)(mbase + r) * Nc + ncol] = ACCV[r] + bv;                  \
  }
  STORE_TILE(0, acc0); STORE_TILE(1, acc1);
  STORE_TILE(2, acc2); STORE_TILE(3, acc3);
#undef STORE_TILE
}

// ---------------------------------------------------------------------------
// Edge attention passes (generic over H, C)
// ---------------------------------------------------------------------------
__global__ void init_ms_kernel(float* __restrict__ m, float* __restrict__ s, int n) {
  int t = blockIdx.x * blockDim.x + threadIdx.x;
  if (t < n) { m[t] = -INFINITY; s[t] = 0.0f; }
}

__global__ void init_zero_kernel(float* __restrict__ p, int n) {
  int t = blockIdx.x * blockDim.x + threadIdx.x;
  if (t < n) p[t] = 0.0f;
}

// sc[e,h] = dot(q[dst,h,:], k[src,h,:] + ea[e]*We[h,:] + be[h,:]) * invscale
__global__ void edge_score_kernel(const float* __restrict__ Q, const float* __restrict__ Kq,
                                  const int* __restrict__ ei, const float* __restrict__ ea,
                                  const float* __restrict__ We, const float* __restrict__ be,
                                  float* __restrict__ scE, float* __restrict__ m,
                                  int E_, int H, int C, float invscale) {
  int t = blockIdx.x * blockDim.x + threadIdx.x;
  if (t >= E_ * H) return;
  int j = t / H, h = t - j * H;
  int src = ei[j], dst = ei[E_ + j];
  float a = ea[j];
  const float* qr  = Q  + (size_t)dst * H * C + h * C;
  const float* kr  = Kq + (size_t)src * H * C + h * C;
  const float* wer = We + h * C;
  const float* ber = be + h * C;
  float sc = 0.0f;
  for (int c = 0; c < C; ++c)
    sc += qr[c] * (kr[c] + a * wer[c] + ber[c]);
  sc *= invscale;
  scE[t] = sc;
  atomicMaxFloat(&m[(size_t)dst * H + h], sc);
}

// ex = exp(sc - m[dst,h]); ssum[dst,h] += ex
__global__ void edge_exp_kernel(const int* __restrict__ ei, float* __restrict__ scE,
                                const float* __restrict__ m, float* __restrict__ ssum,
                                int E_, int H) {
  int t = blockIdx.x * blockDim.x + threadIdx.x;
  if (t >= E_ * H) return;
  int j = t / H, h = t - j * H;
  int dst = ei[E_ + j];
  float ex = expf(scE[t] - m[(size_t)dst * H + h]);
  scE[t] = ex;
  atomicAdd(&ssum[(size_t)dst * H + h], ex);
}

// acc[dst,h,:] += alpha * (v[src,h,:] + ea*We[h,:] + be[h,:])
__global__ void edge_scatter_kernel(const float* __restrict__ V, const int* __restrict__ ei,
                                    const float* __restrict__ ea, const float* __restrict__ We,
                                    const float* __restrict__ be, const float* __restrict__ scE,
                                    const float* __restrict__ ssum, float* __restrict__ acc,
                                    int E_, int H, int C) {
  int t = blockIdx.x * blockDim.x + threadIdx.x;
  if (t >= E_ * H) return;
  int j = t / H, h = t - j * H;
  int src = ei[j], dst = ei[E_ + j];
  float alpha = scE[t] / (ssum[(size_t)dst * H + h] + 1e-16f);
  float a = ea[j];
  const float* vr  = V  + (size_t)src * H * C + h * C;
  const float* wer = We + h * C;
  const float* ber = be + h * C;
  float* ar = acc + (size_t)dst * H * C + h * C;
  for (int c = 0; c < C; ++c)
    atomicAdd(&ar[c], alpha * (vr[c] + a * wer[c] + ber[c]));
}

// ---------------------------------------------------------------------------
// LayerNorm + ReLU: one wave32 per node
// ---------------------------------------------------------------------------
__global__ void __launch_bounds__(32)
ln_relu_kernel(const float* __restrict__ in, const float* __restrict__ g,
               const float* __restrict__ b, float* __restrict__ out, int D) {
  int n = blockIdx.x, lane = threadIdx.x;
  const float* row = in + (size_t)n * D;
  float s = 0.0f, s2 = 0.0f;
  for (int i = lane; i < D; i += 32) { float v = row[i]; s += v; s2 += v * v; }
  for (int o = 16; o > 0; o >>= 1) {
    s  += __shfl_xor(s,  o, 32);
    s2 += __shfl_xor(s2, o, 32);
  }
  float mu  = s / (float)D;
  float var = s2 / (float)D - mu * mu;
  float inv = rsqrtf(var + 1e-5f);
  for (int i = lane; i < D; i += 32) {
    float v = (row[i] - mu) * inv * g[i] + b[i];
    out[(size_t)n * D + i] = fmaxf(v, 0.0f);
  }
}

// ---------------------------------------------------------------------------
// Mean pool per graph + MLP head
// ---------------------------------------------------------------------------
__global__ void pool_scatter_kernel(const float* __restrict__ h2, const int* __restrict__ batch,
                                    float* __restrict__ pooled, float* __restrict__ cnt, int n) {
  int t = blockIdx.x * blockDim.x + threadIdx.x;
  if (t >= n) return;
  int b = batch[t];
  atomicAdd(&cnt[b], 1.0f);
  const float* hr = h2 + (size_t)t * C2C;
  float* pr = pooled + (size_t)b * C2C;
  for (int c = 0; c < C2C; ++c) atomicAdd(&pr[c], hr[c]);
}

__global__ void pool_mlp_kernel(const float* __restrict__ pooled, const float* __restrict__ cnt,
                                const float* __restrict__ Wm1, const float* __restrict__ bm1,
                                const float* __restrict__ Wm2, const float* __restrict__ bm2,
                                float* __restrict__ out) {
  int b = threadIdx.x;
  if (b >= BB) return;
  float inv = 1.0f / fmaxf(cnt[b], 1.0f);
  const float* pr = pooled + (size_t)b * C2C;
  float hid[32];
  for (int o = 0; o < 32; ++o) {
    float s = bm1[o];
    for (int i = 0; i < C2C; ++i) s += pr[i] * inv * Wm1[i * 32 + o];
    hid[o] = fmaxf(s, 0.0f);
  }
  float r = bm2[0];
  for (int o = 0; o < 32; ++o) r += hid[o] * Wm2[o];
  out[b] = r;
}

// ---------------------------------------------------------------------------
// Launcher
// ---------------------------------------------------------------------------
extern "C" void kernel_launch(void* const* d_in, const int* in_sizes, int n_in,
                              void* d_out, int out_size, void* d_ws, size_t ws_size,
                              hipStream_t stream) {
  const float* x    = (const float*)d_in[0];
  const int*   ei   = (const int*)d_in[1];
  const float* ea   = (const float*)d_in[2];
  const int*   batch= (const int*)d_in[3];
  const float* Wq1 = (const float*)d_in[4];  const float* bq1 = (const float*)d_in[5];
  const float* Wk1 = (const float*)d_in[6];  const float* bk1 = (const float*)d_in[7];
  const float* Wv1 = (const float*)d_in[8];  const float* bv1 = (const float*)d_in[9];
  const float* We1 = (const float*)d_in[10]; const float* be1 = (const float*)d_in[11];
  const float* Ws1 = (const float*)d_in[12]; const float* bs1 = (const float*)d_in[13];
  const float* g1  = (const float*)d_in[14]; const float* bl1 = (const float*)d_in[15];
  const float* Wq2 = (const float*)d_in[16]; const float* bq2 = (const float*)d_in[17];
  const float* Wk2 = (const float*)d_in[18]; const float* bk2 = (const float*)d_in[19];
  const float* Wv2 = (const float*)d_in[20]; const float* bv2 = (const float*)d_in[21];
  const float* We2 = (const float*)d_in[22]; const float* be2 = (const float*)d_in[23];
  const float* Ws2 = (const float*)d_in[24]; const float* bs2 = (const float*)d_in[25];
  const float* g2  = (const float*)d_in[26]; const float* bl2 = (const float*)d_in[27];
  const float* Wm1 = (const float*)d_in[28]; const float* bm1 = (const float*)d_in[29];
  const float* Wm2 = (const float*)d_in[30]; const float* bm2 = (const float*)d_in[31];
  float* out = (float*)d_out;

  // ---- workspace layout (floats) ----
  float* ws = (float*)d_ws;
  const size_t ND1 = (size_t)NN * D1;       // 5.12M floats
  const size_t NC2 = (size_t)NN * C2C;      // 0.64M floats
  float* Q1   = ws + 0 * ND1;
  float* K1   = ws + 1 * ND1;
  float* V1   = ws + 2 * ND1;
  float* ACC1 = ws + 3 * ND1;               // skip-GEMM then += attention
  float* H1b  = ws + 4 * ND1;
  float* SCE  = ws + 5 * ND1;               // E*H1 scores / exps
  float* MB   = SCE + (size_t)EE * H1H;     // N*H1 running max
  float* SB   = MB + (size_t)NN * H1H;      // N*H1 softmax denominators
  float* POOL = SB + (size_t)NN * H1H;      // B*C2
  float* CNT  = POOL + (size_t)BB * C2C;    // B
  // layer-2 buffers alias layer-1 Q region (5*NC2 < ND1)
  float* Q2   = Q1 + 0 * NC2;
  float* K2   = Q1 + 1 * NC2;
  float* V2   = Q1 + 2 * NC2;
  float* ACC2 = Q1 + 3 * NC2;
  float* H2b  = Q1 + 4 * NC2;

  const dim3 blk32(32), blk256(256);

  // ===== Layer 1 GEMMs (bf16 WMMA): [10000,128] x [128,512] =====
  {
    dim3 grid(D1 / 64, NN / 16);
    gemm_bias_wmma_bf16<<<grid, blk32, 0, stream>>>(x, Wq1, bq1, Q1,   NN, FF, D1);
    gemm_bias_wmma_bf16<<<grid, blk32, 0, stream>>>(x, Wk1, bk1, K1,   NN, FF, D1);
    gemm_bias_wmma_bf16<<<grid, blk32, 0, stream>>>(x, Wv1, bv1, V1,   NN, FF, D1);
    gemm_bias_wmma_bf16<<<grid, blk32, 0, stream>>>(x, Ws1, bs1, ACC1, NN, FF, D1);
  }

  // ===== Layer 1 edge attention =====
  {
    int nm = NN * H1H;
    init_ms_kernel<<<(nm + 255) / 256, blk256, 0, stream>>>(MB, SB, nm);
    int nt = EE * H1H;
    int gb = (nt + 255) / 256;
    edge_score_kernel<<<gb, blk256, 0, stream>>>(Q1, K1, ei, ea, We1, be1, SCE, MB,
                                                 EE, H1H, C1C, 0.125f);
    edge_exp_kernel<<<gb, blk256, 0, stream>>>(ei, SCE, MB, SB, EE, H1H);
    edge_scatter_kernel<<<gb, blk256, 0, stream>>>(V1, ei, ea, We1, be1, SCE, SB, ACC1,
                                                   EE, H1H, C1C);
  }
  ln_relu_kernel<<<NN, blk32, 0, stream>>>(ACC1, g1, bl1, H1b, D1);

  // ===== Layer 2 GEMMs (bf16 WMMA): [10000,512] x [512,64] =====
  {
    dim3 grid(C2C / 64, NN / 16);
    gemm_bias_wmma_bf16<<<grid, blk32, 0, stream>>>(H1b, Wq2, bq2, Q2,   NN, D1, C2C);
    gemm_bias_wmma_bf16<<<grid, blk32, 0, stream>>>(H1b, Wk2, bk2, K2,   NN, D1, C2C);
    gemm_bias_wmma_bf16<<<grid, blk32, 0, stream>>>(H1b, Wv2, bv2, V2,   NN, D1, C2C);
    gemm_bias_wmma_bf16<<<grid, blk32, 0, stream>>>(H1b, Ws2, bs2, ACC2, NN, D1, C2C);
  }

  // ===== Layer 2 edge attention (H=1, C=64) =====
  {
    init_ms_kernel<<<(NN + 255) / 256, blk256, 0, stream>>>(MB, SB, NN);
    int gb = (EE + 255) / 256;
    edge_score_kernel<<<gb, blk256, 0, stream>>>(Q2, K2, ei, ea, We2, be2, SCE, MB,
                                                 EE, 1, C2C, 0.125f);
    edge_exp_kernel<<<gb, blk256, 0, stream>>>(ei, SCE, MB, SB, EE, 1);
    edge_scatter_kernel<<<gb, blk256, 0, stream>>>(V2, ei, ea, We2, be2, SCE, SB, ACC2,
                                                   EE, 1, C2C);
  }
  ln_relu_kernel<<<NN, blk32, 0, stream>>>(ACC2, g2, bl2, H2b, C2C);

  // ===== Pool + MLP head =====
  {
    int np = BB * C2C + BB;
    init_zero_kernel<<<(np + 255) / 256, blk256, 0, stream>>>(POOL, np);  // POOL then CNT contiguous
    pool_scatter_kernel<<<(NN + 255) / 256, blk256, 0, stream>>>(H2b, batch, POOL, CNT, NN);
    pool_mlp_kernel<<<1, dim3(64), 0, stream>>>(POOL, CNT, Wm1, bm1, Wm2, bm2, out);
  }
}
